// LocallyGroupedSelfAttention_51238959841397
// MI455X (gfx1250) — matrix operand order repaired and checked by reading, not compiled, not run on hardware
//
#include <hip/hip_runtime.h>
#include <hip/hip_bf16.h>

// ---------------------------------------------------------------------------
// Locally-grouped self-attention (Twins-SVT), MI455X / gfx1250, wave32 WMMA.
// B=8, H=W=113, C=512, NH=8, hd=64, ws=7 -> 17x17=289 windows of N=49 tokens.
// Pipeline: [QKV GEMM bf16-wmma] -> [per-(b,win,head) wave attention, wmma]
//           -> [proj GEMM bf16-wmma, A staged via TDM tensor_load_to_lds]
// ---------------------------------------------------------------------------

typedef __attribute__((ext_vector_type(8)))  float        v8f;
typedef __attribute__((ext_vector_type(16))) __bf16       v16bf;
typedef __attribute__((ext_vector_type(8)))  __bf16       v8bf;
typedef __attribute__((ext_vector_type(4)))  __bf16       v4bf;
typedef __attribute__((ext_vector_type(4)))  unsigned int u32x4;
typedef __attribute__((ext_vector_type(8)))  int          i32x8;
typedef __attribute__((ext_vector_type(4)))  int          i32x4;

#define B_   8
#define H_   113
#define W_   113
#define C_   512
#define NH_  8
#define HD_  64
#define WS_  7
#define HW_  (H_ * W_)          // 12769
#define MTOT (B_ * HW_)         // 102152
#define NWIN (17 * 17)          // 289 windows per image

static __device__ __forceinline__ v16bf cat8(v8bf lo, v8bf hi) {
  return __builtin_shufflevector(lo, hi, 0,1,2,3,4,5,6,7,8,9,10,11,12,13,14,15);
}

// ---------------------------------------------------------------------------
// TDM: async-load a [tileRows=128] x [tileK=32] bf16 tile (row stride K elems)
// from global to LDS. D# per CDNA5 ISA:
//   group0: count=1 | lds_addr | global_addr[56:0] | type=2
//   group1: data_size=2B, tensor_dim0=K, tensor_dim1=remRows (OOB rows -> 0),
//           tile_dim0=32, tile_dim1=128, tensor_dim0_stride=K
// group2/3 zero (<=2D). Tracked by TENSORcnt.
// ---------------------------------------------------------------------------
__device__ __forceinline__ void tdm_load_a_tile(const __bf16* gtile,
                                                unsigned ldsOff, int K,
                                                int remRows) {
  unsigned long long ga = (unsigned long long)(uintptr_t)gtile;
  u32x4 g0;
  g0.x = 1u;                                        // count=1
  g0.y = ldsOff;                                    // lds_addr (bytes)
  g0.z = (unsigned)(ga & 0xFFFFFFFFull);            // global_addr[31:0]
  g0.w = (unsigned)((ga >> 32) & 0x01FFFFFFull)     // global_addr[56:32]
         | (2u << 30);                              // type=2 ("image")
  i32x8 g1;
  g1[0] = (1 << 16);                                // data_size=1 -> 2 bytes
  g1[1] = (K & 0xFFFF) << 16;                       // tensor_dim0[15:0]
  g1[2] = ((K >> 16) & 0xFFFF) | ((remRows & 0xFFFF) << 16);
  g1[3] = ((remRows >> 16) & 0xFFFF) | (32 << 16);  // tile_dim0=32
  g1[4] = 128;                                      // tile_dim1=128
  g1[5] = K;                                        // tensor_dim0_stride[31:0]
  g1[6] = 0;
  g1[7] = 0;
  i32x4 z4 = {0, 0, 0, 0};
#if __clang_major__ >= 23
  i32x8 z8 = {0, 0, 0, 0, 0, 0, 0, 0};
  __builtin_amdgcn_tensor_load_to_lds(g0, g1, z4, z4, z8, 0);
#else
  __builtin_amdgcn_tensor_load_to_lds(g0, g1, z4, z4, 0);
#endif
}

// ---------------------------------------------------------------------------
// Tiled GEMM: Out[M,N] = A[M,K] * W[K,N] (+bias). A fp32 or bf16; W fp32.
// Block tile 128x128, k-slab 32. 256 threads = 8 waves; wave w computes the
// 16-row strip [16w,16w+16) x 128 cols => 8 wmma per k-slab per wave.
// A_BF16 path: A tile staged by TDM (double-buffered, copy/compute overlap).
// Bs stored transposed [n][k] so each lane's 32x16 B operand is one
// contiguous 32B LDS read.
// ---------------------------------------------------------------------------
template <bool A_BF16, bool OUT_F32>
__global__ __launch_bounds__(256)
void gemm_wmma_kernel(const void* __restrict__ Ap,
                      const float* __restrict__ Wt,
                      const float* __restrict__ bias,
                      void* __restrict__ Out,
                      int M, int K, int N) {
  constexpr bool USE_TDM = A_BF16;
  __shared__ __bf16 As[2][128][32];   // 16 KB (buf 1 used only in TDM mode)
  __shared__ __bf16 Bs[128][32];      //  8 KB  (Bs[n][k] = W[k0+k][n0+n])

  const int m0 = blockIdx.x * 128;
  const int n0 = blockIdx.y * 128;
  const int tid   = threadIdx.x;
  const int wave  = tid >> 5;
  const int lane  = tid & 31;
  const int lhalf = lane >> 4;     // 0: lanes 0-15, 1: lanes 16-31
  const int l16   = lane & 15;

  v8f acc[8];
  const v8f zf = {};
#pragma unroll
  for (int i = 0; i < 8; ++i) acc[i] = zf;

  if constexpr (USE_TDM) {
    if (wave == 0) {  // kick off first A tile DMA
      tdm_load_a_tile((const __bf16*)Ap + (size_t)m0 * K,
                      (unsigned)(uintptr_t)&As[0][0][0], K, M - m0);
    }
  }

  int buf = 0;
  for (int k0 = 0; k0 < K; k0 += 32) {
    if constexpr (!USE_TDM) {
      // ---- stage A (128x32) fp32 -> bf16 LDS, 16 elems / thread ----
#pragma unroll
      for (int it = 0; it < 4; ++it) {
        int e4 = (it * 256 + tid) * 4;       // [0,4096)
        int r  = e4 >> 5;
        int c  = e4 & 31;
        int gm = m0 + r;
        v4bf pk = {};
        if (gm < M) {
          float4 f = *(const float4*)((const float*)Ap + (size_t)gm * K + k0 + c);
          pk = v4bf{(__bf16)f.x, (__bf16)f.y, (__bf16)f.z, (__bf16)f.w};
        }
        *(v4bf*)&As[0][r][c] = pk;
      }
    }
    // ---- stage B (32x128) transposed -> Bs[n][k] ----
#pragma unroll
    for (int it = 0; it < 4; ++it) {
      int ci  = it * 256 + tid;            // [0,1024) chunks of 4 n
      int kr  = ci >> 5;
      int nc4 = (ci & 31) * 4;
      float4 wv = *(const float4*)(Wt + (size_t)(k0 + kr) * N + n0 + nc4);
      Bs[nc4 + 0][kr] = (__bf16)wv.x;
      Bs[nc4 + 1][kr] = (__bf16)wv.y;
      Bs[nc4 + 2][kr] = (__bf16)wv.z;
      Bs[nc4 + 3][kr] = (__bf16)wv.w;
    }
    if constexpr (USE_TDM) {
      if (wave == 0) __builtin_amdgcn_s_wait_tensorcnt(0);  // A tile landed
    }
    __syncthreads();

    if constexpr (USE_TDM) {
      // async-prefetch next A slab into the other buffer while we compute
      if (wave == 0 && k0 + 32 < K) {
        tdm_load_a_tile((const __bf16*)Ap + (size_t)m0 * K + (k0 + 32),
                        (unsigned)(uintptr_t)&As[buf ^ 1][0][0], K, M - m0);
      }
    } else if (k0 + 32 < K) {
      // prefetch next k-slab into L2 while we compute
      int gm = m0 + ((tid * 4) >> 5);
      if (gm < M)
        __builtin_prefetch((const float*)Ap + (size_t)gm * K + k0 + 32, 0, 0);
      __builtin_prefetch(Wt + (size_t)(k0 + 32 + (tid >> 5)) * N + n0 + (tid & 31) * 4, 0, 0);
    }

    // ---- A operand: 16x32 tile, lane M = l16, K = lhalf*16 + [0,16)
    const int kb = lhalf * 16;
    v8bf alo = *(const v8bf*)&As[buf][wave * 16 + l16][kb];
    v8bf ahi = *(const v8bf*)&As[buf][wave * 16 + l16][kb + 8];
    v16bf a  = cat8(alo, ahi);

    // ---- B operand: 32x16 tile, lane N = l16, K = lhalf*16 + [0,16)
    const int koff = lhalf * 16;
#pragma unroll
    for (int n = 0; n < 8; ++n) {
      v16bf b = *(const v16bf*)&Bs[n * 16 + l16][koff];
      acc[n] = __builtin_amdgcn_wmma_f32_16x16x32_bf16(
          false, a, false, b, (short)0, acc[n], false, false);
    }
    __syncthreads();
    if constexpr (USE_TDM) buf ^= 1;
  }

  // ---- store C: VGPR r -> M = r + lhalf*8 (within wave strip), N = l16 ----
  const int gmBase = m0 + wave * 16 + lhalf * 8;
#pragma unroll
  for (int n = 0; n < 8; ++n) {
    int gn = n0 + n * 16 + l16;
    float bv = (OUT_F32 && bias) ? bias[gn] : 0.0f;
#pragma unroll
    for (int r = 0; r < 8; ++r) {
      int gm = gmBase + r;
      if (gm < M) {
        if (OUT_F32)
          ((float*)Out)[(size_t)gm * N + gn] = acc[n][r] + bv;
        else
          ((__bf16*)Out)[(size_t)gm * N + gn] = (__bf16)acc[n][r];
      }
    }
  }
}

// ---------------------------------------------------------------------------
// Windowed attention: one wave per (b, window, head). N=49 tokens padded to
// 64. S = Q K^T (wmma) -> mask+softmax in registers (shfl_xor over 16-lane
// halves; legal since C-layout keeps N in lane%16) -> P to LDS -> O = P V.
// qkv layout: [b*HW + n][ i*512 + head*64 + d ], i in {q,k,v}.
// ---------------------------------------------------------------------------
__global__ __launch_bounds__(32)
void lgsa_attn_kernel(const __bf16* __restrict__ qkv,
                      __bf16* __restrict__ attn_out) {
  __shared__ __bf16 Kb[64][64];   // [token][hd]   (B operand for Q K^T)
  __shared__ __bf16 Vt[64][64];   // [hd][token]   (B operand for P V)
  __shared__ __bf16 Pb[64][64];   // softmax probs [row tok][col tok]

  const int lin  = blockIdx.x;
  const int head = lin & 7;
  const int wg   = lin >> 3;
  const int b    = wg / NWIN;
  const int wid  = wg % NWIN;
  const int wy   = wid / 17, wx = wid % 17;

  const int lane  = threadIdx.x;
  const int lhalf = lane >> 4;
  const int l16   = lane & 15;

  auto tok2n = [&](int t) -> int {   // window token -> flat spatial index, -1 = pad
    if (t >= 49) return -1;
    int r = wy * WS_ + t / WS_;
    int c = wx * WS_ + t % WS_;
    return (r < H_ && c < W_) ? (r * W_ + c) : -1;
  };

  // ---- stage K rows and V^T columns into LDS (2 tokens per lane) ----
  const v8bf z8 = {};
  for (int t = lane; t < 64; t += 32) {
    int n = tok2n(t);
    if (n >= 0) {
      const __bf16* kp = qkv + ((size_t)(b * HW_ + n)) * (3 * C_) + C_ + head * HD_;
      const __bf16* vp = kp + C_;
#pragma unroll
      for (int c = 0; c < 64; c += 8) {
        v8bf kv = *(const v8bf*)(kp + c);
        *(v8bf*)&Kb[t][c] = kv;
        v8bf vv = *(const v8bf*)(vp + c);
#pragma unroll
        for (int e = 0; e < 8; ++e) Vt[c + e][t] = vv[e];
      }
    } else {
#pragma unroll
      for (int c = 0; c < 64; c += 8) {
        *(v8bf*)&Kb[t][c] = z8;
#pragma unroll
        for (int e = 0; e < 8; ++e) Vt[c + e][t] = (__bf16)0.0f;
      }
    }
  }
  __syncthreads();

  const float scale = 0.125f;   // hd^-0.5, hd=64

  // per-lane column pad status for the 4 col tiles (N = 16n + l16)
  int cOOB[4], cPad[4];
#pragma unroll
  for (int n = 0; n < 4; ++n) {
    int ct  = n * 16 + l16;
    cOOB[n] = (ct >= 49);
    cPad[n] = (!cOOB[n] && tok2n(ct) < 0) ? 1 : 0;
  }

  // ---- S = Q K^T, mask, softmax, P -> LDS ----
#pragma unroll
  for (int mc = 0; mc < 4; ++mc) {
    v8f s[4];
    const v8f zf = {};
#pragma unroll
    for (int n = 0; n < 4; ++n) s[n] = zf;

    int rq = mc * 16 + l16;             // A-operand row token (M = lane%16)
    int nq = tok2n(rq);
    const __bf16* qp =
        (nq >= 0) ? qkv + ((size_t)(b * HW_ + nq)) * (3 * C_) + head * HD_ : nullptr;

#pragma unroll
    for (int kk = 0; kk < 2; ++kk) {
      int kb = kk * 32 + lhalf * 16;
      v8bf alo = z8, ahi = z8;
      if (qp) { alo = *(const v8bf*)(qp + kb); ahi = *(const v8bf*)(qp + kb + 8); }
      v16bf a = cat8(alo, ahi);
      int koff = kk * 32 + lhalf * 16;
#pragma unroll
      for (int n = 0; n < 4; ++n) {
        v16bf bm = *(const v16bf*)&Kb[n * 16 + l16][koff];
        s[n] = __builtin_amdgcn_wmma_f32_16x16x32_bf16(
            false, a, false, bm, (short)0, s[n], false, false);
      }
    }

    // mask + row softmax, register-resident (row M = r + lhalf*8)
#pragma unroll
    for (int r = 0; r < 8; ++r) {
      int rt   = mc * 16 + r + lhalf * 8;
      int rPad = (rt < 49 && tok2n(rt) < 0) ? 1 : 0;
      float mx = -3.0e38f;
#pragma unroll
      for (int n = 0; n < 4; ++n) {
        float v = s[n][r] * scale;
        v += cOOB[n] ? -1.0e30f : (cPad[n] != rPad ? -1000.0f : 0.0f);
        s[n][r] = v;
        mx = fmaxf(mx, v);
      }
#pragma unroll
      for (int off = 1; off < 16; off <<= 1) mx = fmaxf(mx, __shfl_xor(mx, off, 32));
      float sum = 0.0f;
#pragma unroll
      for (int n = 0; n < 4; ++n) {
        float e = __expf(s[n][r] - mx);
        s[n][r] = e;
        sum += e;
      }
#pragma unroll
      for (int off = 1; off < 16; off <<= 1) sum += __shfl_xor(sum, off, 32);
      float inv = 1.0f / sum;
#pragma unroll
      for (int n = 0; n < 4; ++n)
        Pb[rt][n * 16 + l16] = (__bf16)(s[n][r] * inv);
    }
  }
  __syncthreads();

  // ---- O = P V ----
#pragma unroll
  for (int mc = 0; mc < 4; ++mc) {
    v8f o[4];
    const v8f zf = {};
#pragma unroll
    for (int n = 0; n < 4; ++n) o[n] = zf;

#pragma unroll
    for (int kk = 0; kk < 2; ++kk) {
      int kb = kk * 32 + lhalf * 16;
      v8bf alo = *(const v8bf*)&Pb[mc * 16 + l16][kb];
      v8bf ahi = *(const v8bf*)&Pb[mc * 16 + l16][kb + 8];
      v16bf a  = cat8(alo, ahi);
      int koff = kk * 32 + lhalf * 16;
#pragma unroll
      for (int n = 0; n < 4; ++n) {
        v16bf bm = *(const v16bf*)&Vt[n * 16 + l16][koff];   // Vt[hd][token]
        o[n] = __builtin_amdgcn_wmma_f32_16x16x32_bf16(
            false, a, false, bm, (short)0, o[n], false, false);
      }
    }
#pragma unroll
    for (int r = 0; r < 8; ++r) {
      int rt  = mc * 16 + r + lhalf * 8;
      int nsp = tok2n(rt);
      if (nsp >= 0) {
        __bf16* op = attn_out + ((size_t)(b * HW_ + nsp)) * C_ + head * HD_;
#pragma unroll
        for (int n = 0; n < 4; ++n) op[n * 16 + l16] = (__bf16)o[n][r];
      }
    }
  }
}

// ---------------------------------------------------------------------------
// Host launcher. ws layout: [ qkv bf16 : MTOT*1536 ][ attn bf16 : MTOT*512 ]
// ---------------------------------------------------------------------------
extern "C" void kernel_launch(void* const* d_in, const int* in_sizes, int n_in,
                              void* d_out, int out_size, void* d_ws, size_t ws_size,
                              hipStream_t stream) {
  const float* x      = (const float*)d_in[0];
  const float* w_qkv  = (const float*)d_in[1];
  const float* w_proj = (const float*)d_in[2];
  const float* b_proj = (const float*)d_in[3];
  // d_in[4]=h, d_in[5]=w are compile-time constants here.

  __bf16* qkv  = (__bf16*)d_ws;
  __bf16* attn = (__bf16*)((char*)d_ws + (size_t)MTOT * (3 * C_) * sizeof(__bf16));

  // 1) QKV projection: [M,512] x [512,1536] -> bf16
  {
    dim3 grid((MTOT + 127) / 128, (3 * C_) / 128);
    gemm_wmma_kernel<false, false><<<grid, 256, 0, stream>>>(
        (const void*)x, w_qkv, nullptr, (void*)qkv, MTOT, C_, 3 * C_);
  }
  // 2) windowed attention: one wave per (b, window, head)
  {
    dim3 grid(B_ * NWIN * NH_);
    lgsa_attn_kernel<<<grid, 32, 0, stream>>>(qkv, attn);
  }
  // 3) output projection + bias: [M,512] x [512,512] -> fp32 d_out (TDM-staged A)
  {
    dim3 grid((MTOT + 127) / 128, C_ / 128);
    gemm_wmma_kernel<true, true><<<grid, 256, 0, stream>>>(
        (const void*)attn, w_proj, b_proj, d_out, MTOT, C_, C_);
  }
}
